// HiLo_48129403519638
// MI455X (gfx1250) — matrix operand order repaired and checked by reading, not compiled, not run on hardware
//
#include <hip/hip_runtime.h>
#include <hip/hip_bf16.h>

// ---------------------------------------------------------------------------
// HiLo attention for MI455X (gfx1250, wave32, WMMA 16x16x32 f16).
// Heavy math -> v_wmma_f32_16x16x32_f16; staging -> global_load_async_to_lds
// with double-buffered LDS tiles (ASYNCcnt pipelining).
// ---------------------------------------------------------------------------

typedef __attribute__((ext_vector_type(16))) _Float16 v16h;
typedef __attribute__((ext_vector_type(8)))  _Float16 v8h;
typedef __attribute__((ext_vector_type(8)))  float    v8f;

static constexpr int   BATCH    = 4;
static constexpr int   CDIM     = 512;
static constexpr int   SPATIAL  = 64 * 64;           // per batch
static constexpr int   NTOK     = BATCH * SPATIAL;   // 16384
static constexpr int   NPOOL    = BATCH * 1024;      // 4096 pooled tokens
static constexpr float SCALE    = 0.125f;            // 64^-0.5

__device__ __forceinline__ v8f wmma_f16(v16h a, v16h b, v8f c) {
  return __builtin_amdgcn_wmma_f32_16x16x32_f16(
      /*neg_a=*/false, a, /*neg_b=*/false, b,
      /*c_mod=*/(short)0, c, /*reuse_a=*/false, /*reuse_b=*/false);
}

// Async DMA: 16 bytes global -> LDS, no VGPR data path (ASYNCcnt tracked).
__device__ __forceinline__ void async_copy_b128(void* lds_dst, const void* gsrc) {
  const unsigned lds = (unsigned)(uintptr_t)lds_dst;   // low 32 bits = LDS offset
  asm volatile("global_load_async_to_lds_b128 %0, %1, off"
               :: "v"(lds), "v"(gsrc) : "memory");
}
__device__ __forceinline__ void async_wait0() {
  asm volatile("s_wait_asynccnt 0x0" ::: "memory");
}

// A-matrix fragment (16x32 f16, row-major source, stride in halves).
// lanes 0-15: row M=lane, K={0..7,16..23}; lanes 16-31: K={8..15,24..31}.
__device__ __forceinline__ v16h load_frag_a(const _Float16* base, int stride, int lane) {
  const int r = lane & 15, hs = lane >> 4;
  const _Float16* p = base + r * stride + hs * 8;
  v8h lo = *(const v8h*)(p);
  v8h hi = *(const v8h*)(p + 16);
  v16h out;
#pragma unroll
  for (int i = 0; i < 8; ++i) { out[i] = lo[i]; out[i + 8] = hi[i]; }
  return out;
}

// B-matrix fragment (32x16 f16). Source stored [N][K]; lanes 0-15:
// K={0..7,16..23}; lanes 16-31: K={8..15,24..31}: two 16B ds_load_b128.
__device__ __forceinline__ v16h load_frag_b(const _Float16* base, int stride, int lane) {
  const int c = lane & 15, hs = lane >> 4;
  const _Float16* p = base + c * stride + hs * 8;
  v8h lo = *(const v8h*)(p);
  v8h hi = *(const v8h*)(p + 16);
  v16h out;
#pragma unroll
  for (int i = 0; i < 8; ++i) { out[i] = lo[i]; out[i + 8] = hi[i]; }
  return out;
}

// ---------------------------------------------------------------------------
// NCHW fp32 -> token-major f16  (xh[t][c]),  LDS tile transpose.
// ---------------------------------------------------------------------------
__global__ __launch_bounds__(256) void nchw_to_tok_f16(
    const float* __restrict__ x, _Float16* __restrict__ xh) {
  __shared__ float tile[32][33];
  const int tx = threadIdx.x & 31, ty = threadIdx.x >> 5;
  const int b = blockIdx.z, c0 = blockIdx.y * 32, p0 = blockIdx.x * 32;
#pragma unroll
  for (int s = 0; s < 4; ++s) {
    int c = ty + s * 8;
    tile[c][tx] = x[((b * CDIM + c0 + c) * SPATIAL) + p0 + tx];
  }
  __syncthreads();
#pragma unroll
  for (int s = 0; s < 4; ++s) {
    int p = ty + s * 8;
    xh[(size_t)(b * SPATIAL + p0 + p) * CDIM + c0 + tx] = (_Float16)tile[tx][p];
  }
}

// ---------------------------------------------------------------------------
// 2x2 avg pool over NCHW fp32 -> pooled f16 [B*1024][512]
// ---------------------------------------------------------------------------
__global__ __launch_bounds__(256) void pool_f16(
    const float* __restrict__ x, _Float16* __restrict__ pooled) {
  int id = blockIdx.x * 256 + threadIdx.x;   // 4*512*1024
  int gw = id & 31, gh = (id >> 5) & 31, c = (id >> 10) & 511, b = id >> 19;
  const float* xp = x + (((size_t)(b * CDIM + c) * 64 + 2 * gh) * 64 + 2 * gw);
  float v = 0.25f * (xp[0] + xp[1] + xp[64] + xp[65]);
  pooled[(size_t)(b * 1024 + gh * 32 + gw) * CDIM + c] = (_Float16)v;
}

// ---------------------------------------------------------------------------
// Weight prep: W[KxN] fp32 -> Wt[NxK] f16 (one-off, tiny).
// ---------------------------------------------------------------------------
__global__ __launch_bounds__(256) void wt_transpose_f16(
    const float* __restrict__ W, _Float16* __restrict__ Wt, int K, int N) {
  const int id = blockIdx.x * 256 + threadIdx.x;
  if (id >= K * N) return;
  const int n = id % N, k = id / N;
  Wt[(size_t)n * K + k] = (_Float16)W[(size_t)k * N + n];
}

// ---------------------------------------------------------------------------
// Tiled WMMA GEMM:  out[MxN] = A[MxK](f16,row-major) * Bt[NxK](f16)
// Block: 256 thr (8 waves), tile 64x128x32.  M%64==0, N%128==0, K%32==0.
// Double-buffered async DMA staging: DMA of tile k+1 overlaps WMMA on tile k.
// MODE 1: f16 row-major out.  MODE 2: fp32 NCHW scatter + bias + cOff.
// ---------------------------------------------------------------------------
template <int MODE>
__global__ __launch_bounds__(256) void gemm_wmma_kernel(
    const _Float16* __restrict__ A, const _Float16* __restrict__ Bt,
    void* __restrict__ outp, const float* __restrict__ bias,
    int M, int N, int K, int cOff) {
  __shared__ __attribute__((aligned(16))) _Float16 As[2][64 * 32];    // [m][k]
  __shared__ __attribute__((aligned(16))) _Float16 Bs[2][128 * 32];   // [n][k]

  const int tid = threadIdx.x, lane = tid & 31, wave = tid >> 5;
  const int wm = wave & 1, wn = wave >> 1;
  const int m0 = blockIdx.y * 64, n0 = blockIdx.x * 128;

  const int ar = tid >> 2, ac = (tid & 3) * 8;   // A: 64 rows x 32k, 16B each
  const int br = tid >> 1, bc = (tid & 1) * 16;  // B: 128 rows x 32k, 2x16B/row

  v8f acc[2][2] = {};

  // prologue: DMA first tiles into buffer 0
  async_copy_b128(&As[0][ar * 32 + ac], &A[(size_t)(m0 + ar) * K + ac]);
  async_copy_b128(&Bs[0][br * 32 + bc], &Bt[(size_t)(n0 + br) * K + bc]);

  int cur = 0;
  for (int k0 = 0; k0 < K; k0 += 32, cur ^= 1) {
    async_wait0();          // tile `cur` DMA (issued last iteration) complete
    __syncthreads();        // ... for every wave in the block
    const int kn = k0 + 32;
    if (kn < K) {           // prefetch next tile into the other buffer
      async_copy_b128(&As[cur ^ 1][ar * 32 + ac],
                      &A[(size_t)(m0 + ar) * K + kn + ac]);
      async_copy_b128(&Bs[cur ^ 1][br * 32 + bc],
                      &Bt[(size_t)(n0 + br) * K + kn + bc]);
    }
    v16h bf0 = load_frag_b(&Bs[cur][(wn * 32) * 32], 32, lane);
    v16h bf1 = load_frag_b(&Bs[cur][(wn * 32 + 16) * 32], 32, lane);
#pragma unroll
    for (int i = 0; i < 2; ++i) {
      v16h af = load_frag_a(&As[cur][(wm * 32 + i * 16) * 32], 32, lane);
      acc[i][0] = wmma_f16(af, bf0, acc[i][0]);
      acc[i][1] = wmma_f16(af, bf1, acc[i][1]);
    }
  }

  const int cc = lane & 15, hs = lane >> 4;
#pragma unroll
  for (int i = 0; i < 2; ++i) {
    const int rbase = m0 + wm * 32 + i * 16 + 8 * hs;   // multiple of 8
#pragma unroll
    for (int j = 0; j < 2; ++j) {
      const int colg = n0 + wn * 32 + j * 16 + cc;
      if (MODE == 1) {
#pragma unroll
        for (int e = 0; e < 8; ++e)
          ((_Float16*)outp)[(size_t)(rbase + e) * N + colg] =
              (_Float16)acc[i][j][e];
      } else {
        // 8 accum elements = 8 consecutive tokens of one output channel.
        const int bb = rbase >> 12, p = rbase & 4095;
        float* dst =
            (float*)outp + ((size_t)(bb * CDIM + cOff + colg)) * SPATIAL + p;
        const float bv = bias[colg];
        const float4 lo4 = make_float4(acc[i][j][0] + bv, acc[i][j][1] + bv,
                                       acc[i][j][2] + bv, acc[i][j][3] + bv);
        const float4 hi4 = make_float4(acc[i][j][4] + bv, acc[i][j][5] + bv,
                                       acc[i][j][6] + bv, acc[i][j][7] + bv);
        *(float4*)(dst) = lo4;
        *(float4*)(dst + 4) = hi4;
      }
    }
  }
}

// ---------------------------------------------------------------------------
// Hi-Fi window attention: N=4 tokens/window, tiny -> VALU (vectorized b128 IO).
// qkv f16 [16384][768], col = which*256 + head*64 + d.  out hi16 [16384][256].
// ---------------------------------------------------------------------------
__global__ __launch_bounds__(128) void attn_hi_kernel(
    const _Float16* __restrict__ qkv, _Float16* __restrict__ hi) {
  const int id = blockIdx.x * 128 + threadIdx.x;     // 65536 = B*G*heads*N
  const int n = id & 3, head = (id >> 2) & 3, g = (id >> 4) & 1023, b = id >> 14;
  const int gh = g >> 5, gw = g & 31;
  int tok[4];
#pragma unroll
  for (int m = 0; m < 4; ++m)
    tok[m] = b * SPATIAL + (2 * gh + (m >> 1)) * 64 + 2 * gw + (m & 1);
  const int qc = head * 64;

  v8h q8[8];
  const _Float16* qp = qkv + (size_t)tok[n] * 768 + qc;
#pragma unroll
  for (int dv = 0; dv < 8; ++dv) q8[dv] = *(const v8h*)(qp + dv * 8);

  float s[4];
#pragma unroll
  for (int m = 0; m < 4; ++m) {
    const _Float16* kp = qkv + (size_t)tok[m] * 768 + 256 + qc;
    float acc = 0.f;
#pragma unroll
    for (int dv = 0; dv < 8; ++dv) {
      const v8h kv = *(const v8h*)(kp + dv * 8);
#pragma unroll
      for (int jj = 0; jj < 8; ++jj) acc += (float)q8[dv][jj] * (float)kv[jj];
    }
    s[m] = acc * SCALE;
  }
  float mx = fmaxf(fmaxf(s[0], s[1]), fmaxf(s[2], s[3]));
  float l = 0.f;
#pragma unroll
  for (int m = 0; m < 4; ++m) { s[m] = __expf(s[m] - mx); l += s[m]; }
  const float inv = 1.f / l;

  _Float16* op = hi + (size_t)tok[n] * 256 + qc;
#pragma unroll
  for (int dv = 0; dv < 8; ++dv) {
    float o8[8] = {};
#pragma unroll
    for (int m = 0; m < 4; ++m) {
      const v8h vv =
          *(const v8h*)(qkv + (size_t)tok[m] * 768 + 512 + qc + dv * 8);
#pragma unroll
      for (int jj = 0; jj < 8; ++jj) o8[jj] += s[m] * (float)vv[jj];
    }
    v8h ov;
#pragma unroll
    for (int jj = 0; jj < 8; ++jj) ov[jj] = (_Float16)(o8[jj] * inv);
    *(v8h*)(op + dv * 8) = ov;
  }
}

// ---------------------------------------------------------------------------
// Lo-Fi attention: flash-style WMMA.  Per (b, head): Q 4096x64 vs K/V 1024x64.
// Block = 8 waves, each wave owns 16 Q rows. Double-buffered K (async DMA)
// and V^T (vectorized load + LDS scatter) staging per 32-key chunk.
// ---------------------------------------------------------------------------
__global__ __launch_bounds__(256) void attn_lo_kernel(
    const _Float16* __restrict__ Q,    // [16384][256]
    const _Float16* __restrict__ KV,   // [4096][512]: which*256 + head*64 + d
    _Float16* __restrict__ O) {        // [16384][256]
  __shared__ __attribute__((aligned(16))) _Float16 Ks[2][32 * 64];  // [key][d]
  __shared__ __attribute__((aligned(16))) _Float16 Vt[2][64 * 32];  // [d][key]
  __shared__ __attribute__((aligned(16))) _Float16 Ps[8 * 16 * 32];

  const int tid = threadIdx.x, lane = tid & 31, wave = tid >> 5;
  const int b = blockIdx.z, head = blockIdx.y;
  const int q0 = b * SPATIAL + blockIdx.x * 128 + wave * 16;
  const int qc = head * 64;
  const int cc = lane & 15, hs = lane >> 4;

  const int kr = tid >> 3, kc = (tid & 7) * 8;           // K stage map
  const int vkey = tid & 31, vdg = (tid >> 5) * 8;       // V stage map

  auto stage = [&](int buf, int kb) {
    async_copy_b128(&Ks[buf][kr * 64 + kc],
                    &KV[(size_t)(b * 1024 + kb + kr) * 512 + qc + kc]);
    const v8h vv =
        *(const v8h*)&KV[(size_t)(b * 1024 + kb + vkey) * 512 + 256 + qc + vdg];
#pragma unroll
    for (int j = 0; j < 8; ++j) Vt[buf][(vdg + j) * 32 + vkey] = vv[j];
  };

  // Preload Q fragments for this wave's 16 rows (d = 0..63 -> two A frags).
  const v16h qa0 = load_frag_a(Q + (size_t)q0 * 256 + qc, 256, lane);
  const v16h qa1 = load_frag_a(Q + (size_t)q0 * 256 + qc + 32, 256, lane);

  v8f o0 = {}, o1 = {}, o2 = {}, o3 = {};
  float mrow[8], lrow[8];
#pragma unroll
  for (int i = 0; i < 8; ++i) { mrow[i] = -1e30f; lrow[i] = 0.f; }

  _Float16* Pw = &Ps[wave * 512];

  stage(0, 0);                          // prologue
  int cur = 0;
  for (int kb = 0; kb < 1024; kb += 32, cur ^= 1) {
    async_wait0();
    __syncthreads();                    // chunk `cur` fully staged
    if (kb + 32 < 1024) stage(cur ^ 1, kb + 32);

    // S = Q * K^T  (two 16-key subtiles, K-dim = 64 split in two WMMAs)
    v8f s0 = {}, s1 = {};
    s0 = wmma_f16(qa0, load_frag_b(&Ks[cur][0], 64, lane), s0);
    s0 = wmma_f16(qa1, load_frag_b(&Ks[cur][32], 64, lane), s0);
    s1 = wmma_f16(qa0, load_frag_b(&Ks[cur][16 * 64], 64, lane), s1);
    s1 = wmma_f16(qa1, load_frag_b(&Ks[cur][16 * 64 + 32], 64, lane), s1);

    // Online softmax in C-fragment layout (row = e + 8*hs, col = lane&15).
#pragma unroll
    for (int i = 0; i < 8; ++i) {
      float a = s0[i] * SCALE, c2 = s1[i] * SCALE;
      float rmax = fmaxf(a, c2);
#pragma unroll
      for (int msk = 8; msk; msk >>= 1)
        rmax = fmaxf(rmax, __shfl_xor(rmax, msk, 16));
      const float mnew = fmaxf(mrow[i], rmax);
      const float corr = __expf(mrow[i] - mnew);
      mrow[i] = mnew;
      const float p0 = __expf(a - mnew), p1 = __expf(c2 - mnew);
      float rsum = p0 + p1;
#pragma unroll
      for (int msk = 8; msk; msk >>= 1) rsum += __shfl_xor(rsum, msk, 16);
      lrow[i] = lrow[i] * corr + rsum;
      o0[i] *= corr; o1[i] *= corr; o2[i] *= corr; o3[i] *= corr;
      Pw[(i + 8 * hs) * 32 + cc] = (_Float16)p0;        // C-layout -> LDS
      Pw[(i + 8 * hs) * 32 + 16 + cc] = (_Float16)p1;
    }
    // Reload P as an A fragment (16x32) and accumulate O += P * V.
    const v16h pf = load_frag_a(Pw, 32, lane);
    o0 = wmma_f16(pf, load_frag_b(&Vt[cur][0], 32, lane), o0);
    o1 = wmma_f16(pf, load_frag_b(&Vt[cur][16 * 32], 32, lane), o1);
    o2 = wmma_f16(pf, load_frag_b(&Vt[cur][32 * 32], 32, lane), o2);
    o3 = wmma_f16(pf, load_frag_b(&Vt[cur][48 * 32], 32, lane), o3);
  }

#pragma unroll
  for (int i = 0; i < 8; ++i) {
    const float inv = 1.f / lrow[i];
    const int row = q0 + i + 8 * hs;
    _Float16* op = O + (size_t)row * 256 + qc;
    op[cc]      = (_Float16)(o0[i] * inv);
    op[16 + cc] = (_Float16)(o1[i] * inv);
    op[32 + cc] = (_Float16)(o2[i] * inv);
    op[48 + cc] = (_Float16)(o3[i] * inv);
  }
}

// ---------------------------------------------------------------------------
extern "C" void kernel_launch(void* const* d_in, const int* in_sizes, int n_in,
                              void* d_out, int out_size, void* d_ws, size_t ws_size,
                              hipStream_t stream) {
  const float* x        = (const float*)d_in[0];
  const float* h_qkv_w  = (const float*)d_in[1];
  const float* h_proj_w = (const float*)d_in[2];
  const float* h_proj_b = (const float*)d_in[3];
  const float* l_q_w    = (const float*)d_in[4];
  const float* l_kv_w   = (const float*)d_in[5];
  const float* l_proj_w = (const float*)d_in[6];
  const float* l_proj_b = (const float*)d_in[7];
  float* out = (float*)d_out;

  // workspace layout (f16 intermediates + pre-transposed f16 weights)
  char* ws = (char*)d_ws;
  size_t off = 0;
  auto alloc = [&](size_t bytes) {
    void* p = ws + off;
    off += (bytes + 255) & ~(size_t)255;
    return p;
  };
  _Float16* xh16     = (_Float16*)alloc((size_t)NTOK * 512 * 2);
  _Float16* pooled16 = (_Float16*)alloc((size_t)NPOOL * 512 * 2);
  _Float16* qkv16    = (_Float16*)alloc((size_t)NTOK * 768 * 2);
  _Float16* qlo16    = (_Float16*)alloc((size_t)NTOK * 256 * 2);
  _Float16* kv16     = (_Float16*)alloc((size_t)NPOOL * 512 * 2);
  _Float16* hi16     = (_Float16*)alloc((size_t)NTOK * 256 * 2);
  _Float16* lo16     = (_Float16*)alloc((size_t)NTOK * 256 * 2);
  _Float16* h_qkv_wt  = (_Float16*)alloc((size_t)512 * 768 * 2);
  _Float16* l_q_wt    = (_Float16*)alloc((size_t)512 * 256 * 2);
  _Float16* l_kv_wt   = (_Float16*)alloc((size_t)512 * 512 * 2);
  _Float16* h_proj_wt = (_Float16*)alloc((size_t)256 * 256 * 2);
  _Float16* l_proj_wt = (_Float16*)alloc((size_t)256 * 256 * 2);

  // 0) one-off weight transposes (fp32 [KxN] -> f16 [NxK])
  wt_transpose_f16<<<(512 * 768) / 256, 256, 0, stream>>>(h_qkv_w, h_qkv_wt, 512, 768);
  wt_transpose_f16<<<(512 * 256) / 256, 256, 0, stream>>>(l_q_w, l_q_wt, 512, 256);
  wt_transpose_f16<<<(512 * 512) / 256, 256, 0, stream>>>(l_kv_w, l_kv_wt, 512, 512);
  wt_transpose_f16<<<(256 * 256) / 256, 256, 0, stream>>>(h_proj_w, h_proj_wt, 256, 256);
  wt_transpose_f16<<<(256 * 256) / 256, 256, 0, stream>>>(l_proj_w, l_proj_wt, 256, 256);

  // 1) layout conversion + pooling
  nchw_to_tok_f16<<<dim3(128, 16, BATCH), 256, 0, stream>>>(x, xh16);
  pool_f16<<<(BATCH * 1024 * CDIM) / 256, 256, 0, stream>>>(x, pooled16);

  // 2) input projections (WMMA GEMMs, double-buffered async staging)
  gemm_wmma_kernel<1><<<dim3(768 / 128, NTOK / 64), 256, 0, stream>>>(
      xh16, h_qkv_wt, qkv16, nullptr, NTOK, 768, 512, 0);
  gemm_wmma_kernel<1><<<dim3(256 / 128, NTOK / 64), 256, 0, stream>>>(
      xh16, l_q_wt, qlo16, nullptr, NTOK, 256, 512, 0);
  gemm_wmma_kernel<1><<<dim3(512 / 128, NPOOL / 64), 256, 0, stream>>>(
      pooled16, l_kv_wt, kv16, nullptr, NPOOL, 512, 512, 0);

  // 3) attention
  attn_hi_kernel<<<65536 / 128, 128, 0, stream>>>(qkv16, hi16);
  attn_lo_kernel<<<dim3(32, 4, BATCH), 256, 0, stream>>>(qlo16, kv16, lo16);

  // 4) output projections fused with bias + NHWC->NCHW scatter + concat
  gemm_wmma_kernel<2><<<dim3(256 / 128, NTOK / 64), 256, 0, stream>>>(
      hi16, h_proj_wt, out, h_proj_b, NTOK, 256, 256, /*cOff=*/0);
  gemm_wmma_kernel<2><<<dim3(256 / 128, NTOK / 64), 256, 0, stream>>>(
      lo16, l_proj_wt, out, l_proj_b, NTOK, 256, 256, /*cOff=*/256);
}
